// EncoderModel_14147622273027
// MI455X (gfx1250) — matrix operand order repaired and checked by reading, not compiled, not run on hardware
//
#include <hip/hip_runtime.h>

typedef unsigned short u16;
typedef __attribute__((ext_vector_type(16))) __bf16 v16bf;
typedef __attribute__((ext_vector_type(8)))  __bf16 v8bf;
typedef __attribute__((ext_vector_type(8)))  float  v8f;

// Problem constants (from reference)
#define BB    64          // batch
#define TT    12          // timesteps
#define HH    6400        // hidden
#define G3H   (3 * HH)    // 19200
#define IN0   200
#define IN0P  224         // padded to multiple of 32
#define BH    (BB * HH)   // 409600
#define KS    4           // K-split factor for the big (K=6400) GEMMs

__device__ __forceinline__ u16 f32_to_bf16(float f) {
  unsigned u = __float_as_uint(f);
  u += 0x7FFFu + ((u >> 16) & 1u);   // round-to-nearest-even
  return (u16)(u >> 16);
}

// ---------- generic fp32 -> bf16 convert ----------
__global__ void k_cvt_bf16(const float* __restrict__ src, u16* __restrict__ dst, long n) {
  long i = (long)blockIdx.x * blockDim.x + threadIdx.x;
  long stride = (long)gridDim.x * blockDim.x;
  for (; i < n; i += stride) dst[i] = f32_to_bf16(src[i]);
}

// ---------- W_ih0 (19200 x 200) -> bf16 padded (19200 x 224) ----------
__global__ void k_pack_wih0(const float* __restrict__ src, u16* __restrict__ dst) {
  long n = (long)G3H * IN0P;
  long i = (long)blockIdx.x * blockDim.x + threadIdx.x;
  long stride = (long)gridDim.x * blockDim.x;
  for (; i < n; i += stride) {
    int r = (int)(i / IN0P), c = (int)(i % IN0P);
    dst[i] = (c < IN0) ? f32_to_bf16(src[(long)r * IN0 + c]) : (u16)0;
  }
}

// ---------- inputs (B,T,IN0) fp32 -> x_pad (T*B, 224) bf16 ----------
__global__ void k_pack_x(const float* __restrict__ src, u16* __restrict__ dst) {
  long n = (long)TT * BB * IN0P;
  long i = (long)blockIdx.x * blockDim.x + threadIdx.x;
  long stride = (long)gridDim.x * blockDim.x;
  for (; i < n; i += stride) {
    int row = (int)(i / IN0P), c = (int)(i % IN0P);
    int t = row / BB, b = row % BB;
    dst[i] = (c < IN0) ? f32_to_bf16(src[((long)b * TT + t) * IN0 + c]) : (u16)0;
  }
}

// ---------- WMMA GEMM (K-split): Cpart[z][M][N] = A[M][kz..] * W[N][kz..]^T ----------
// grid.x = N/128, grid.y = M/64, grid.z = k-split parts, block = 256 (8 wave32s).
// Each wave: N strip of 16, four stacked 16x16 M tiles, K stepped by 32.
__global__ void __launch_bounds__(256)
k_gemm_bf16(float* __restrict__ C, long cPartStride,
            const u16* __restrict__ A, const u16* __restrict__ W,
            int kLen, int lda, int ldw, int ldc)
{
  const int wave = threadIdx.x >> 5;
  const int lane = threadIdx.x & 31;
  const int ln = lane & 15;        // N col / M row within tile
  const int lh = lane >> 4;        // half-wave selector
  const int n0 = blockIdx.x * 128 + wave * 16;
  const int m0 = blockIdx.y * 64;
  const int kBase = blockIdx.z * kLen;

  v8f acc[4] = {v8f{0.f}, v8f{0.f}, v8f{0.f}, v8f{0.f}};

  // B-matrix: lane ln holds column n0+ln, 16 contiguous K-halves starting at lh*16
  const u16* wrow = W + (long)(n0 + ln) * ldw + kBase + lh * 16;
  // A-matrix: lane holds row m&15; chunk0 at khalf lh*16, chunk1 at +8
  const u16* arow = A + (long)(m0 + ln) * lda + kBase + lh * 16;

  for (int k = 0; k < kLen; k += 32) {
    // Weight stream is 245.8 MB per GEMM (> 192 MB L2): pure streaming, keep
    // it non-temporal so activations/partials stay resident in cache.
    v16bf bm = __builtin_nontemporal_load((const v16bf*)(wrow + k));
#pragma unroll
    for (int mt = 0; mt < 4; ++mt) {
      const u16* ap = arow + (long)mt * 16 * lda + k;
      v8bf alo = *(const v8bf*)(ap);
      v8bf ahi = *(const v8bf*)(ap + 8);
      v16bf am;
#pragma unroll
      for (int i = 0; i < 8; ++i) { am[i] = alo[i]; am[i + 8] = ahi[i]; }
      acc[mt] = __builtin_amdgcn_wmma_f32_16x16x32_bf16(
          false, am, false, bm, (short)0, acc[mt], false, false);
    }
  }

  float* Cz = C + (long)blockIdx.z * cPartStride;
#pragma unroll
  for (int mt = 0; mt < 4; ++mt) {
    float* cp = Cz + (long)(m0 + mt * 16 + lh * 8) * ldc + n0 + ln;
#pragma unroll
    for (int r = 0; r < 8; ++r) cp[(long)r * ldc] = acc[mt][r];
  }
}

// ---------- GRU gate pointwise (sums K-split partials, adds both biases) ----------
__global__ void k_gru_gate(const float* __restrict__ gi, long giStride, int giParts,
                           const float* __restrict__ gh, long ghStride, int ghParts,
                           const float* __restrict__ bih, const float* __restrict__ bhh,
                           float* __restrict__ hbuf,     // B*H fp32, in/out (in place OK)
                           u16* __restrict__ hbf_out,    // B*H bf16 for next GEMMs
                           float* __restrict__ out2)     // optional extra fp32 dest
{
  int idx = blockIdx.x * blockDim.x + threadIdx.x;
  if (idx >= BH) return;
  int b = idx / HH, j = idx % HH;
  long o = (long)b * G3H + j;

  float gr = bih[j], gz = bih[HH + j], gn = bih[2 * HH + j];
  for (int p = 0; p < giParts; ++p) {
    const float* g = gi + (long)p * giStride + o;
    gr += g[0]; gz += g[HH]; gn += g[2 * HH];
  }
  float hr = bhh[j], hz = bhh[HH + j], hn = bhh[2 * HH + j];
  for (int p = 0; p < ghParts; ++p) {
    const float* g = gh + (long)p * ghStride + o;
    hr += g[0]; hz += g[HH]; hn += g[2 * HH];
  }

  float r  = 1.f / (1.f + __expf(-(gr + hr)));
  float z  = 1.f / (1.f + __expf(-(gz + hz)));
  float nn = tanhf(gn + r * hn);
  float hp = hbuf[idx];
  float hnew = (1.f - z) * nn + z * hp;
  hbuf[idx] = hnew;
  hbf_out[idx] = f32_to_bf16(hnew);
  if (out2) out2[idx] = hnew;
}

extern "C" void kernel_launch(void* const* d_in, const int* in_sizes, int n_in,
                              void* d_out, int out_size, void* d_ws, size_t ws_size,
                              hipStream_t stream) {
  (void)in_sizes; (void)n_in; (void)out_size; (void)ws_size;
  const float* inputs = (const float*)d_in[0];
  const float* W_ih0  = (const float*)d_in[1];
  const float* W_hh0  = (const float*)d_in[2];
  const float* b_ih0  = (const float*)d_in[3];
  const float* b_hh0  = (const float*)d_in[4];
  const float* W_ih1  = (const float*)d_in[5];
  const float* W_hh1  = (const float*)d_in[6];
  const float* b_ih1  = (const float*)d_in[7];
  const float* b_hh1  = (const float*)d_in[8];
  float* out = (float*)d_out;

  const long PART = (long)BB * G3H;   // floats per partial C buffer

  // carve workspace (256B aligned slabs)
  char* wsp = (char*)d_ws;
  auto carve = [&](size_t bytes) -> char* {
    char* p = wsp; wsp += (bytes + 255) & ~(size_t)255; return p;
  };
  u16*   wih0p = (u16*)carve((size_t)G3H * IN0P * 2);     //  8.6 MB
  u16*   whh0b = (u16*)carve((size_t)G3H * HH * 2);       // 245.8 MB
  u16*   wih1b = (u16*)carve((size_t)G3H * HH * 2);       // 245.8 MB
  u16*   whh1b = (u16*)carve((size_t)G3H * HH * 2);       // 245.8 MB
  u16*   xpad  = (u16*)carve((size_t)TT * BB * IN0P * 2);
  float* gi0   = (float*)carve((size_t)TT * PART * 4);    // 59 MB (single part)
  float* gi1p  = (float*)carve((size_t)KS * PART * 4);    // 19.7 MB
  float* ghp   = (float*)carve((size_t)KS * PART * 4);    // 19.7 MB (reused gh0/gh1)
  u16*   h0seq = (u16*)carve((size_t)(TT + 1) * BH * 2);  // slot 0 = zeros
  u16*   h1seq = (u16*)carve((size_t)(TT + 1) * BH * 2);
  float* h0f   = (float*)carve((size_t)BH * 4);
  float* h1f   = (float*)carve((size_t)BH * 4);

  // 1) weight / input conversion to bf16
  k_pack_wih0<<<2048, 256, 0, stream>>>(W_ih0, wih0p);
  k_cvt_bf16 <<<4096, 256, 0, stream>>>(W_hh0, whh0b, (long)G3H * HH);
  k_cvt_bf16 <<<4096, 256, 0, stream>>>(W_ih1, wih1b, (long)G3H * HH);
  k_cvt_bf16 <<<4096, 256, 0, stream>>>(W_hh1, whh1b, (long)G3H * HH);
  k_pack_x   <<<512,  256, 0, stream>>>(inputs, xpad);

  // 2) zero-init recurrent state (every call; graph replays this)
  hipMemsetAsync(h0seq, 0, (size_t)BH * 2, stream);
  hipMemsetAsync(h1seq, 0, (size_t)BH * 2, stream);
  hipMemsetAsync(h0f,   0, (size_t)BH * 4, stream);
  hipMemsetAsync(h1f,   0, (size_t)BH * 4, stream);

  // 3) layer-0 input projections for all timesteps: (T*B,224) x (19200,224)^T
  k_gemm_bf16<<<dim3(G3H / 128, (TT * BB) / 64, 1), 256, 0, stream>>>(
      gi0, 0, xpad, wih0p, IN0P, IN0P, IN0P, G3H);

  dim3 gBig(G3H / 128, 1, KS);          // 600 blocks per big GEMM
  const int gateBlocks = (BH + 255) / 256;
  const int kPart = HH / KS;            // 1600, multiple of 32

  // 4) sequential recurrence
  for (int t = 0; t < TT; ++t) {
    // layer 0: gh0 = h0_{t} @ W_hh0^T   (KS partials)
    k_gemm_bf16<<<gBig, 256, 0, stream>>>(ghp, PART, h0seq + (size_t)t * BH, whh0b,
                                          kPart, HH, HH, G3H);
    k_gru_gate<<<gateBlocks, 256, 0, stream>>>(
        gi0 + (size_t)t * PART, 0, 1, ghp, PART, KS, b_ih0, b_hh0,
        h0f, h0seq + (size_t)(t + 1) * BH, nullptr);
    // layer 1 input projection: gi1 = h0_{t+1} @ W_ih1^T   (KS partials)
    k_gemm_bf16<<<gBig, 256, 0, stream>>>(gi1p, PART, h0seq + (size_t)(t + 1) * BH, wih1b,
                                          kPart, HH, HH, G3H);
    // layer 1 recurrent: gh1 = h1_{t} @ W_hh1^T   (KS partials)
    k_gemm_bf16<<<gBig, 256, 0, stream>>>(ghp, PART, h1seq + (size_t)t * BH, whh1b,
                                          kPart, HH, HH, G3H);
    k_gru_gate<<<gateBlocks, 256, 0, stream>>>(
        gi1p, PART, KS, ghp, PART, KS, b_ih1, b_hh1,
        h1f, h1seq + (size_t)(t + 1) * BH, out + (size_t)t * BH);
  }

  // 5) hidden = stack([h0_last, h1_last])
  hipMemcpyAsync(out + (size_t)TT * BH,      h0f, (size_t)BH * 4,
                 hipMemcpyDeviceToDevice, stream);
  hipMemcpyAsync(out + (size_t)TT * BH + BH, h1f, (size_t)BH * 4,
                 hipMemcpyDeviceToDevice, stream);
}
